// OverlapCrossAttention_43705587204172
// MI455X (gfx1250) — matrix operand order repaired and checked by reading, not compiled, not run on hardware
//
#include <hip/hip_runtime.h>
#include <hip/hip_fp16.h>
#include <math.h>

// ---------------------------------------------------------------------------
// Overlap cross-attention for MI455X (gfx1250, wave32, WMMA 16x16x32 f16).
//  - Flash-style streaming softmax: score tensor (1.2 GB) never hits memory.
//  - GEMM waves compute 16x256 panels (16 WMMA tiles per A load) so the big
//    f32 activations are read from HBM once (kv twice): ~0.6 GB total traffic
//    -> ~26 us at 23.3 TB/s; f16 WMMA keeps compute under that.
//  - Attention: one 512-thread block per (b,h); K/V^T staged to LDS in
//    double-buffered 64-key chunks via GLOBAL_LOAD_ASYNC_TO_LDS_B128
//    (ASYNCcnt + split barrier), shared by all 16 query-tile waves.
// ---------------------------------------------------------------------------

typedef _Float16 f16;
typedef _Float16 v16h __attribute__((ext_vector_type(16)));
typedef _Float16 v8h  __attribute__((ext_vector_type(8)));
typedef float    v8f  __attribute__((ext_vector_type(8)));
typedef float    v4f  __attribute__((ext_vector_type(4)));
typedef int      vsi4 __attribute__((vector_size(16)));  // b128 payload type

#define NB      256   // batch*windows
#define NH      8
#define DH      32
#define DMODEL  256
#define NQ      256   // M*M
#define NKK     576   // Mo*Mo
#define CHUNK   64    // keys per LDS chunk
#define NCHUNK  9     // 576/64

// ---- CDNA5 async global->LDS path (guarded; fallback = load + ds_store) ---
#ifndef USE_ASYNC_LDS
# if defined(__has_builtin)
#  if __has_builtin(__builtin_amdgcn_global_load_async_to_lds_b128)
#   define USE_ASYNC_LDS 1
#  endif
# endif
#endif
#ifndef USE_ASYNC_LDS
# define USE_ASYNC_LDS 0
#endif

#if USE_ASYNC_LDS
# if __has_builtin(__builtin_amdgcn_s_wait_asynccnt)
#  define WAIT_ASYNC(n) __builtin_amdgcn_s_wait_asynccnt(n)
# else
#  define WAIT_ASYNC(n) asm volatile("s_wait_asynccnt %0" ::"i"(n) : "memory")
# endif
static __device__ __forceinline__ void async_copy16(const f16* gsrc, f16* ldst) {
  // signature (from hipcc diagnostic): (int4 AS1*, int4 AS3*, imm, imm)
  __builtin_amdgcn_global_load_async_to_lds_b128(
      (__attribute__((address_space(1))) vsi4*)(void*)gsrc,
      (__attribute__((address_space(3))) vsi4*)(void*)ldst, 0, 0);
}
#else
# define WAIT_ASYNC(n)
static __device__ __forceinline__ void async_copy16(const f16* gsrc, f16* ldst) {
  *(uint4*)ldst = *(const uint4*)gsrc;
}
#endif

static __device__ __forceinline__ v8f wmma_f16(v16h a, v16h b, v8f c) {
  // D = A(16x32 f16) * B(32x16 f16) + C(16x16 f32)
  return __builtin_amdgcn_wmma_f32_16x16x32_f16(false, a, false, b, (short)0, c,
                                                false, false);
}

// A-operand (16x32, f16 source). Lane l: row=l%16, g=l>>4;
// e0..15 = K(16g..16g+15).
static __device__ __forceinline__ v16h load_a_f16(const f16* ap, int g) {
  v8h lo = *(const v8h*)(ap + 16 * g);
  v8h hi = *(const v8h*)(ap + 16 * g + 8);
  return __builtin_shufflevector(lo, hi, 0, 1, 2, 3, 4, 5, 6, 7,
                                 8, 9, 10, 11, 12, 13, 14, 15);
}

static __device__ __forceinline__ v16h load_a_f32(const float* ap, int g) {
  v4f x0 = *(const v4f*)(ap + 16 * g);
  v4f x1 = *(const v4f*)(ap + 16 * g + 4);
  v4f x2 = *(const v4f*)(ap + 16 * g + 8);
  v4f x3 = *(const v4f*)(ap + 16 * g + 12);
  v16h a;
#pragma unroll
  for (int i = 0; i < 4; ++i) {
    a[i]      = (f16)x0[i];
    a[4 + i]  = (f16)x1[i];
    a[8 + i]  = (f16)x2[i];
    a[12 + i] = (f16)x3[i];
  }
  return a;
}

// ------------------------- prep kernels ------------------------------------

// Wt[n*256 + k] = W[k*N + n]  (K input dim fixed at 256)
__global__ void prep_wt(const float* __restrict__ W, f16* __restrict__ Wt,
                        int N, int total) {
  int i = blockIdx.x * 256 + threadIdx.x;
  if (i >= total) return;
  int n = i >> 8;
  int k = i & 255;
  Wt[i] = (f16)W[(size_t)k * N + n];
}

// biasT[h][kcol][qrow] = table[rel[qrow*576 + kcol]*8 + h]
__global__ void prep_bias(const float* __restrict__ table,
                          const int* __restrict__ rel,
                          float* __restrict__ biasT) {
  int i = blockIdx.x * 256 + threadIdx.x;  // over 8*576*256
  if (i >= NH * NKK * NQ) return;
  int q = i & 255;
  int t = i >> 8;
  int kcol = t % NKK;
  int h = t / NKK;
  biasT[i] = table[rel[q * NKK + kcol] * NH + h];
}

// ------------------------- GEMM core ---------------------------------------
// Wave computes a 16-row x (NT*16)-col panel; one A tile feeds NT WMMAs/K-step.
template <int NT, bool AF16>
static __device__ __forceinline__ void gemm_panel(const void* A_,
                                                  const f16* __restrict__ Bt,
                                                  int rowbase, int colbase,
                                                  int lane16, int g,
                                                  v8f acc[NT]) {
  const f16* b0 = Bt + (size_t)(colbase + lane16) * DMODEL + 16 * g;
#pragma unroll
  for (int kk = 0; kk < DMODEL; kk += 32) {
    v16h a;
    if (AF16)
      a = load_a_f16((const f16*)A_ + (size_t)(rowbase + lane16) * DMODEL + kk, g);
    else
      a = load_a_f32((const float*)A_ + (size_t)(rowbase + lane16) * DMODEL + kk, g);
#pragma unroll
    for (int t = 0; t < NT; ++t) {
      v16h b = *(const v16h*)(b0 + (size_t)t * 16 * DMODEL + kk);
      acc[t] = wmma_f16(a, b, acc[t]);
    }
  }
}

// q projection: 16x256 per wave, A (f32) read once; scale folded into output.
__global__ void __launch_bounds__(256)
gemm_qproj(const float* __restrict__ A, const f16* __restrict__ Bt,
           const float* __restrict__ bias, f16* __restrict__ out) {
  const int lane = threadIdx.x & 31, wave = threadIdx.x >> 5;
  const int lane16 = lane & 15, g = lane >> 4;
  const int rowbase = blockIdx.x * 128 + wave * 16;
  v8f acc[16];
  const v8f z = {0, 0, 0, 0, 0, 0, 0, 0};
#pragma unroll
  for (int t = 0; t < 16; ++t) acc[t] = z;
  gemm_panel<16, false>(A, Bt, rowbase, 0, lane16, g, acc);
  const float scale = 0.17677669529663687f;  // Dh^-1/2
#pragma unroll
  for (int t = 0; t < 16; ++t) {
    int col = t * 16 + lane16;
    float bi = bias[col];
#pragma unroll
    for (int r = 0; r < 8; ++r) {
      int row = rowbase + r + 8 * g;
      out[(size_t)row * DMODEL + col] = (f16)((acc[t][r] + bi) * scale);
    }
  }
}

// kv projection: y-block 0 -> K (B,H,Nk,Dh); y-block 1 -> V transposed
// (B,H,Dh,Nk) so attention V^T tiles load contiguously.
__global__ void __launch_bounds__(256)
gemm_kvproj(const float* __restrict__ A, const f16* __restrict__ Bt,
            const float* __restrict__ bias, f16* __restrict__ kf,
            f16* __restrict__ vt) {
  const int lane = threadIdx.x & 31, wave = threadIdx.x >> 5;
  const int lane16 = lane & 15, g = lane >> 4;
  const int rowbase = blockIdx.x * 128 + wave * 16;
  const int colbase = blockIdx.y * 256;
  v8f acc[16];
  const v8f z = {0, 0, 0, 0, 0, 0, 0, 0};
#pragma unroll
  for (int t = 0; t < 16; ++t) acc[t] = z;
  gemm_panel<16, false>(A, Bt, rowbase, colbase, lane16, g, acc);
  if (colbase == 0) {  // K half (uniform per block)
#pragma unroll
    for (int t = 0; t < 16; ++t) {
      int col = t * 16 + lane16;
      float bi = bias[col];
      int hh = col >> 5, d = col & 31;
#pragma unroll
      for (int r = 0; r < 8; ++r) {
        int row = rowbase + r + 8 * g;
        int b = row / NKK, n = row % NKK;
        kf[(((size_t)b * NH + hh) * NKK + n) * DH + d] = (f16)(acc[t][r] + bi);
      }
    }
  } else {  // V half -> transposed
#pragma unroll
    for (int t = 0; t < 16; ++t) {
      int col = colbase + t * 16 + lane16;
      float bi = bias[col];
      int cc = col - DMODEL;
      int hh = cc >> 5, d = cc & 31;
#pragma unroll
      for (int r = 0; r < 8; ++r) {
        int row = rowbase + r + 8 * g;
        int b = row / NKK, n = row % NKK;
        vt[(((size_t)b * NH + hh) * DH + d) * NKK + n] = (f16)(acc[t][r] + bi);
      }
    }
  }
}

// output projection: 16x256 per wave, A (f16) read once, f32 result.
__global__ void __launch_bounds__(256)
gemm_outproj(const f16* __restrict__ A, const f16* __restrict__ Bt,
             const float* __restrict__ bias, float* __restrict__ out) {
  const int lane = threadIdx.x & 31, wave = threadIdx.x >> 5;
  const int lane16 = lane & 15, g = lane >> 4;
  const int rowbase = blockIdx.x * 128 + wave * 16;
  v8f acc[16];
  const v8f z = {0, 0, 0, 0, 0, 0, 0, 0};
#pragma unroll
  for (int t = 0; t < 16; ++t) acc[t] = z;
  gemm_panel<16, true>(A, Bt, rowbase, 0, lane16, g, acc);
#pragma unroll
  for (int t = 0; t < 16; ++t) {
    int col = t * 16 + lane16;
    float bi = bias[col];
#pragma unroll
    for (int r = 0; r < 8; ++r) {
      int row = rowbase + r + 8 * g;
      out[(size_t)row * DMODEL + col] = acc[t][r] + bi;
    }
  }
}

// ------------------------- attention kernel --------------------------------
// One 512-thread block (16 waves) per (b,h); wave = one 16-query tile.
// S^T = K*q^T keeps a fixed query row per lane -> softmax stats are per-lane
// scalars (+one shfl_xor 16); P^T operand needs only a half-wave swap.
// K/V^T double-buffered in LDS (16 KB), async-prefetched one chunk ahead.
__global__ void __launch_bounds__(512)
attn_kernel(const f16* __restrict__ qf, const f16* __restrict__ kf,
            const f16* __restrict__ vt, const float* __restrict__ biasT,
            f16* __restrict__ ao) {
  const int bh = blockIdx.x;  // 0..2047
  const int b = bh >> 3, h = bh & 7;
  const int wave = threadIdx.x >> 5;  // = q tile 0..15
  const int lane = threadIdx.x & 31;
  const int lane16 = lane & 15, g = lane >> 4;
  const int qrow = wave * 16 + lane16;

  __shared__ f16 sk[2][CHUNK * DH];  // keys, row-major (key, d)
  __shared__ f16 sv[2][DH * CHUNK];  // V^T, row-major (d, key)

  // q^T B-operand: lane col = qrow, K elements = head dims 16g..16g+15.
  v16h bq = *(const v16h*)(qf + ((size_t)b * NQ + qrow) * DMODEL + h * DH + 16 * g);

  const f16* kbase = kf + (size_t)bh * NKK * DH;
  const f16* vbase = vt + (size_t)bh * DH * NKK;
  const float* bb = biasT + (size_t)h * NKK * NQ;

  // per-thread staging job: threads 0..255 copy K (one b128 each),
  // 256..511 copy V^T (one b128 each). 512 * 16B = 8 KB = one chunk pair.
  const int tid = threadIdx.x;
  const int vu = tid - 256;
  const int vd = vu >> 3, vj = (vu & 7) * 8;

  auto issue = [&](int c, int buf) {
    int kb = c * CHUNK;
    if (tid < 256)
      async_copy16(kbase + (size_t)kb * DH + tid * 8, &sk[buf][tid * 8]);
    else
      async_copy16(vbase + (size_t)vd * NKK + kb + vj, &sv[buf][vd * CHUNK + vj]);
  };

  v8f o0 = {0, 0, 0, 0, 0, 0, 0, 0}, o1 = {0, 0, 0, 0, 0, 0, 0, 0};
  float mi = -INFINITY, li = 0.0f;

  issue(0, 0);
  for (int c = 0; c < NCHUNK; ++c) {
    const int buf = c & 1;
    if (c + 1 < NCHUNK) {
      issue(c + 1, buf ^ 1);
      WAIT_ASYNC(1);  // chunk c's async loads complete (in-order)
    } else {
      WAIT_ASYNC(0);
    }
    __syncthreads();  // chunk c visible to all 16 waves

#pragma unroll
    for (int sub = 0; sub < 2; ++sub) {
      const int kl = sub * 32;       // key offset within chunk
      const int kg = c * CHUNK + kl; // global key offset
      // ---- scores S^T: two 16-key tiles, one WMMA each (K = Dh = 32)
      v16h ak0 = load_a_f16(&sk[buf][(kl + lane16) * DH], g);
      v16h ak1 = load_a_f16(&sk[buf][(kl + 16 + lane16) * DH], g);
      v8f s0 = {0, 0, 0, 0, 0, 0, 0, 0}, s1 = {0, 0, 0, 0, 0, 0, 0, 0};
      s0 = wmma_f16(ak0, bq, s0);
      s1 = wmma_f16(ak1, bq, s1);
      // lane reg r holds (kcol = kg(+16) + r + 8g, qrow): add gathered bias
#pragma unroll
      for (int r = 0; r < 8; ++r) {
        s0[r] += bb[(size_t)(kg + r + 8 * g) * NQ + qrow];
        s1[r] += bb[(size_t)(kg + 16 + r + 8 * g) * NQ + qrow];
      }
      // ---- streaming softmax (row == lane; pair with lane^16)
      float mx = -INFINITY;
#pragma unroll
      for (int r = 0; r < 8; ++r) mx = fmaxf(mx, fmaxf(s0[r], s1[r]));
      mx = fmaxf(mx, __shfl_xor(mx, 16, 32));
      float mnew = fmaxf(mi, mx);
      float corr = __expf(mi - mnew);
      mi = mnew;
      float rs = 0.0f;
#pragma unroll
      for (int r = 0; r < 8; ++r) {
        s0[r] = __expf(s0[r] - mnew);
        s1[r] = __expf(s1[r] - mnew);
        rs += s0[r] + s1[r];
      }
      rs += __shfl_xor(rs, 16, 32);
      li = li * corr + rs;
#pragma unroll
      for (int r = 0; r < 8; ++r) { o0[r] *= corr; o1[r] *= corr; }
      // ---- P^T B-operand (32 keys x 16 queries) via half-wave swap
      v16h bp;
#pragma unroll
      for (int r = 0; r < 8; ++r) {
        float a = s0[r], bv = s1[r];
        float asw = __shfl_xor(a, 16, 32);
        float bsw = __shfl_xor(bv, 16, 32);
        bp[r]     = (f16)(g ? bsw : a);
        bp[r + 8] = (f16)(g ? bv : asw);
      }
      // ---- O^T += V^T * P^T  (two d-tiles of 16)
      v16h av0 = load_a_f16(&sv[buf][lane16 * CHUNK + kl], g);
      v16h av1 = load_a_f16(&sv[buf][(16 + lane16) * CHUNK + kl], g);
      o0 = wmma_f16(av0, bp, o0);
      o1 = wmma_f16(av1, bp, o1);
    }
    __syncthreads();  // all waves done with buf before it is overwritten
  }

  const float inv = 1.0f / li;
  v8h p0, p1;
#pragma unroll
  for (int r = 0; r < 8; ++r) {
    p0[r] = (f16)(o0[r] * inv);
    p1[r] = (f16)(o1[r] * inv);
  }
  f16* op = ao + ((size_t)b * NQ + qrow) * DMODEL + h * DH;
  *(v8h*)(op + 8 * g)      = p0;
  *(v8h*)(op + 16 + 8 * g) = p1;
}

// ------------------------- launcher ----------------------------------------

extern "C" void kernel_launch(void* const* d_in, const int* in_sizes, int n_in,
                              void* d_out, int out_size, void* d_ws,
                              size_t ws_size, hipStream_t stream) {
  const float* q_w        = (const float*)d_in[0];
  const float* kv_w       = (const float*)d_in[1];
  const float* q_W        = (const float*)d_in[2];
  const float* q_b        = (const float*)d_in[3];
  const float* kv_W       = (const float*)d_in[4];
  const float* kv_b       = (const float*)d_in[5];
  const float* proj_W     = (const float*)d_in[6];
  const float* proj_b     = (const float*)d_in[7];
  const float* bias_table = (const float*)d_in[8];
  const int*   rel_index  = (const int*)d_in[9];

  char* w = (char*)d_ws;
  f16* qWt   = (f16*)w;   w += (size_t)65536 * 2;
  f16* kvWt  = (f16*)w;   w += (size_t)131072 * 2;
  f16* pWt   = (f16*)w;   w += (size_t)65536 * 2;
  float* biasT = (float*)w; w += (size_t)NH * NKK * NQ * 4;
  f16* qf  = (f16*)w;     w += (size_t)NB * NQ * DMODEL * 2;
  f16* kf  = (f16*)w;     w += (size_t)NB * NH * NKK * DH * 2;
  f16* vtp = (f16*)w;     w += (size_t)NB * NH * NKK * DH * 2;
  f16* ao  = (f16*)w;     w += (size_t)NB * NQ * DMODEL * 2;
  (void)ws_size; (void)in_sizes; (void)n_in; (void)out_size;

  prep_wt<<<256, 256, 0, stream>>>(q_W, qWt, 256, 65536);
  prep_wt<<<512, 256, 0, stream>>>(kv_W, kvWt, 512, 131072);
  prep_wt<<<256, 256, 0, stream>>>(proj_W, pWt, 256, 65536);
  prep_bias<<<4608, 256, 0, stream>>>(bias_table, rel_index, biasT);

  gemm_qproj<<<dim3(512), 256, 0, stream>>>(q_w, qWt, q_b, qf);
  gemm_kvproj<<<dim3(1152, 2), 256, 0, stream>>>(kv_w, kvWt, kv_b, kf, vtp);
  attn_kernel<<<2048, 512, 0, stream>>>(qf, kf, vtp, biasT, ao);
  gemm_outproj<<<dim3(512), 256, 0, stream>>>(ao, pWt, proj_b, (float*)d_out);
}
